// DyMuS_39118562132048
// MI455X (gfx1250) — matrix operand order repaired and checked by reading, not compiled, not run on hardware
//
#include <hip/hip_runtime.h>
#include <hip/hip_bf16.h>

// ---------------------------------------------------------------------------
// DyMuS multi-behavior capsule-routing model for MI455X (gfx1250, wave32).
// Matrix work runs on V_WMMA_F32_16X16X32_F16 (f16 in, f32 accumulate).
// ---------------------------------------------------------------------------

typedef __attribute__((ext_vector_type(16))) _Float16 v16h;
typedef __attribute__((ext_vector_type(8)))  float    v8f;

#define BATCH   512
#define EMB     128     // E == H == C
#define CAPD    8
#define NBEH    4
#define VOCAB   50001   // V+1
#define VPAD    50048   // 391 * 128
#define GATES   384     // 3*E

// ---------------------------------------------------------------------------
// WMMA helpers — fragment layouts per CDNA5 ISA 7.12.2 (wave32).
// ---------------------------------------------------------------------------
static __device__ __forceinline__ v8f wmma_f16(v16h a, v16h b, v8f c) {
  return __builtin_amdgcn_wmma_f32_16x16x32_f16(false, a, false, b, (short)0, c,
                                                false, false);
}

// A (16x32 f16): lane<16 -> row m=lane, K {0..7,16..23}; lane>=16 -> K {8..15,24..31}
static __device__ __forceinline__ v16h load_a_frag(const _Float16* a, int ldk, int kbase) {
  const int lane = threadIdx.x & 31;
  const int m = lane & 15, h = lane >> 4;
  const _Float16* r = a + (size_t)m * ldk + kbase;
  v16h f;
#pragma unroll
  for (int i = 0; i < 8; ++i) {
    f[i]     = r[h * 8 + i];
    f[8 + i] = r[16 + h * 8 + i];
  }
  return f;
}

// B (32x16 f16) from an N-major source (row n holds contiguous K):
// lane<16 -> col n=lane, K {0..7,16..23} ; lane>=16 -> K {8..15,24..31}
static __device__ __forceinline__ v16h load_b_frag_nmajor(const _Float16* bm, int ldk, int kbase) {
  const int lane = threadIdx.x & 31;
  const int n = lane & 15, h = lane >> 4;
  const _Float16* r = bm + (size_t)n * ldk + kbase;
  v16h f;
#pragma unroll
  for (int i = 0; i < 8; ++i) {
    f[i]     = r[h * 8 + i];
    f[8 + i] = r[16 + h * 8 + i];
  }
  return f;
}

// C/D (16x16 f32): vgpr r -> row m = r + 8*(lane>=16), col n = lane&15
static __device__ __forceinline__ void store_acc_lds(float* out, int ldn, v8f acc) {
  const int lane = threadIdx.x & 31;
  const int n = lane & 15, h = lane >> 4;
#pragma unroll
  for (int r = 0; r < 8; ++r) out[(size_t)(r + 8 * h) * ldn + n] = acc[r];
}

// ---------------------------------------------------------------------------
// Conversion / init kernels
// ---------------------------------------------------------------------------
__global__ void cvt_f32_to_f16_kernel(const float* __restrict__ src,
                                      _Float16* __restrict__ dst, int n) {
  int i = blockIdx.x * blockDim.x + threadIdx.x;
  if (i < n) dst[i] = (_Float16)src[i];
}

__global__ void cvt_itememb_kernel(const float* __restrict__ src,
                                   _Float16* __restrict__ dst) {
  int i = blockIdx.x * blockDim.x + threadIdx.x;  // over VPAD*128
  int row = i >> 7;
  dst[i] = (row < VOCAB) ? (_Float16)src[i] : (_Float16)0.0f;
}

__global__ void zero_f32_kernel(float* __restrict__ dst, int n) {
  int i = blockIdx.x * blockDim.x + threadIdx.x;
  if (i < n) dst[i] = 0.0f;
}

// ---------------------------------------------------------------------------
// Fused GRU: embedding gather + (x @ WihT) + (h @ WhhT) + gates, per behavior.
// grid = B/16 blocks, 256 threads (8 waves). Wave w owns N-tiles 3w..3w+2.
// ---------------------------------------------------------------------------
__global__ __launch_bounds__(256) void gru_kernel(
    const int* __restrict__ seq, int T,
    const _Float16* __restrict__ itemE,     // VPAD x 128 f16
    const _Float16* __restrict__ wih,       // 384 x 128 f16 (this behavior)
    const _Float16* __restrict__ whh,       // 384 x 128 f16
    const float* __restrict__ bih,          // 384
    const float* __restrict__ bhh,          // 384
    float* __restrict__ e_out,              // (B, 128, NBEH)
    int beh) {
  __shared__ _Float16 xt[16 * 128];
  __shared__ _Float16 h16[16 * 128];
  __shared__ float gi[16 * GATES];
  __shared__ float gh[16 * GATES];

  const int tid  = threadIdx.x;
  const int wave = tid >> 5;
  const int b_base = blockIdx.x * 16;

  const int idx0  = tid * 8;       // this thread owns 8 contiguous (m,e) slots
  const int m_own = idx0 >> 7;
  const int e_own = idx0 & 127;

  float hreg[8];
#pragma unroll
  for (int i = 0; i < 8; ++i) {
    hreg[i] = 0.0f;
    h16[idx0 + i] = (_Float16)0.0f;
  }
  __syncthreads();

  for (int t = 0; t < T; ++t) {
    // -- gather embedding row (item_emb[0] is zero; mask anyway) --
    const int sidx = seq[(size_t)(b_base + m_own) * T + t];
    const _Float16* src = itemE + (size_t)sidx * 128;
#pragma unroll
    for (int i = 0; i < 8; ++i)
      xt[idx0 + i] = (sidx == 0) ? (_Float16)0.0f : src[e_own + i];
    __syncthreads();

    // -- gate GEMMs: gi = x @ WihT, gh = h @ WhhT  (M=16,N=384,K=128) --
#pragma unroll
    for (int ti = 0; ti < 3; ++ti) {
      const int nt = wave * 3 + ti;
      v8f ai = {}, ah = {};
#pragma unroll
      for (int ks = 0; ks < 4; ++ks) {
        v16h af = load_a_frag(xt, 128, ks * 32);
        v16h hf = load_a_frag(h16, 128, ks * 32);
        v16h bi = load_b_frag_nmajor(wih + (size_t)nt * 16 * 128, 128, ks * 32);
        v16h bh = load_b_frag_nmajor(whh + (size_t)nt * 16 * 128, 128, ks * 32);
        ai = wmma_f16(af, bi, ai);
        ah = wmma_f16(hf, bh, ah);
      }
      store_acc_lds(gi + nt * 16, GATES, ai);
      store_acc_lds(gh + nt * 16, GATES, ah);
    }
    __syncthreads();

    // -- elementwise gates + state update --
#pragma unroll
    for (int i = 0; i < 8; ++i) {
      const int e = e_own + i;
      const float ir  = gi[m_own * GATES + e]       + bih[e];
      const float iz  = gi[m_own * GATES + 128 + e] + bih[128 + e];
      const float inn = gi[m_own * GATES + 256 + e] + bih[256 + e];
      const float hr  = gh[m_own * GATES + e]       + bhh[e];
      const float hz  = gh[m_own * GATES + 128 + e] + bhh[128 + e];
      const float hn  = gh[m_own * GATES + 256 + e] + bhh[256 + e];
      const float r = 1.0f / (1.0f + __expf(-(ir + hr)));
      const float z = 1.0f / (1.0f + __expf(-(iz + hz)));
      const float n = tanhf(inn + r * hn);
      const float h = (1.0f - z) * n + z * hreg[i];
      hreg[i] = h;
      h16[m_own * 128 + e] = (_Float16)h;
    }
    __syncthreads();
  }

  // e layout (B, H, NBEH)
#pragma unroll
  for (int i = 0; i < 8; ++i)
    e_out[((size_t)(b_base + m_own) * 128 + e_own + i) * NBEH + beh] = hreg[i];
}

// ---------------------------------------------------------------------------
// u[b,h,c,d] = sum_t e[b,h,t] * W[h,t,c,d]     (thread per (b,h,c); 8 d's)
// ---------------------------------------------------------------------------
__global__ void u_kernel(const float* __restrict__ e,
                         const float* __restrict__ W,
                         float* __restrict__ u) {
  const size_t idx = (size_t)blockIdx.x * blockDim.x + threadIdx.x;  // B*H*C
  const int b = (int)(idx >> 14);
  const int h = (int)((idx >> 7) & 127);
  const int c = (int)(idx & 127);
  const float* ev = e + ((size_t)b * 128 + h) * NBEH;
  float acc[CAPD];
#pragma unroll
  for (int d = 0; d < CAPD; ++d) acc[d] = 0.0f;
#pragma unroll
  for (int t = 0; t < NBEH; ++t) {
    const float* w = W + (((size_t)h * NBEH + t) * 128 + c) * CAPD;
    const float ef = ev[t];
#pragma unroll
    for (int d = 0; d < CAPD; ++d) acc[d] += ef * w[d];
  }
  float* out = u + idx * CAPD;
#pragma unroll
  for (int d = 0; d < CAPD; ++d) out[d] = acc[d];
}

// ---------------------------------------------------------------------------
// c = softmax(b) over capsule axis C (row length 128).  grid = B*H, 128 thr.
// ---------------------------------------------------------------------------
__global__ void softmaxC_kernel(const float* __restrict__ bL,
                                float* __restrict__ cC) {
  __shared__ float red[128];
  const int tid = threadIdx.x;
  const size_t base = (size_t)blockIdx.x * 128;
  const float x = bL[base + tid];
  red[tid] = x;
  __syncthreads();
  for (int s = 64; s >= 1; s >>= 1) {
    if (tid < s) red[tid] = fmaxf(red[tid], red[tid + s]);
    __syncthreads();
  }
  const float mx = red[0];
  __syncthreads();
  const float ex = __expf(x - mx);
  red[tid] = ex;
  __syncthreads();
  for (int s = 64; s >= 1; s >>= 1) {
    if (tid < s) red[tid] += red[tid + s];
    __syncthreads();
  }
  cC[base + tid] = ex / red[0];
}

// ---------------------------------------------------------------------------
// v[b,c,:] = alpha * sum_h c[b,h,c]*u[b,h,c,:]; v_ = ||v||*w_scale + bias.
// grid = B*C, 64 threads.
// ---------------------------------------------------------------------------
__global__ void v_kernel(const float* __restrict__ cC,
                         const float* __restrict__ u,
                         const float* __restrict__ alpha,
                         const float* __restrict__ w_scale,
                         const float* __restrict__ bias_vec,
                         float* __restrict__ v,
                         float* __restrict__ vnorm,
                         _Float16* __restrict__ vnorm16) {
  __shared__ float red[64][CAPD];
  const int tid = threadIdx.x;
  const int b = blockIdx.x >> 7;
  const int c = blockIdx.x & 127;
  float acc[CAPD];
#pragma unroll
  for (int d = 0; d < CAPD; ++d) acc[d] = 0.0f;
  for (int h = tid; h < 128; h += 64) {
    const size_t base = ((size_t)b * 128 + h) * 128 + c;
    const float cc = cC[base];
    const float* uu = u + base * CAPD;
#pragma unroll
    for (int d = 0; d < CAPD; ++d) acc[d] += cc * uu[d];
  }
#pragma unroll
  for (int d = 0; d < CAPD; ++d) red[tid][d] = acc[d];
  __syncthreads();
  for (int s = 32; s >= 1; s >>= 1) {
    if (tid < s)
#pragma unroll
      for (int d = 0; d < CAPD; ++d) red[tid][d] += red[tid + s][d];
    __syncthreads();
  }
  if (tid == 0) {
    const float a0 = alpha[0];
    float s2 = 0.0f;
#pragma unroll
    for (int d = 0; d < CAPD; ++d) {
      const float val = a0 * red[0][d];
      v[(size_t)blockIdx.x * CAPD + d] = val;
      s2 += val * val;
    }
    const float nv = sqrtf(s2) * w_scale[c] + bias_vec[c];
    vnorm[blockIdx.x] = nv;
    vnorm16[blockIdx.x] = (_Float16)nv;
  }
}

// ---------------------------------------------------------------------------
// logits = A(M x128, f16) @ Brows^T  (Brows: N x 128 f16, N-major).
// grid = (M/16, ceil(N/128)); 8 waves, one 16x16 output tile each.
// ---------------------------------------------------------------------------
__global__ __launch_bounds__(256) void gemm_aBt_kernel(
    const _Float16* __restrict__ A,
    const _Float16* __restrict__ Brows,
    float* __restrict__ out, long out_ld, int n_limit) {
  const int wave = threadIdx.x >> 5;
  const int lane = threadIdx.x & 31;
  const int m_base = blockIdx.x * 16;
  const size_t n_base = (size_t)blockIdx.y * 128 + wave * 16;
  v8f acc = {};
#pragma unroll
  for (int ks = 0; ks < 4; ++ks) {
    v16h af = load_a_frag(A + (size_t)m_base * 128, 128, ks * 32);
    v16h bf = load_b_frag_nmajor(Brows + n_base * 128, 128, ks * 32);
    acc = wmma_f16(af, bf, acc);
  }
  const long n_g = (long)n_base + (lane & 15);
  const int h = lane >> 4;
  if (n_g < n_limit) {
#pragma unroll
    for (int r = 0; r < 8; ++r)
      out[(size_t)(m_base + r + 8 * h) * out_ld + n_g] = acc[r];
  }
}

// ---------------------------------------------------------------------------
// Row softmax stats over the logits (only first `ncols` columns count).
// ---------------------------------------------------------------------------
__global__ void softmax_stats_kernel(const float* __restrict__ l, long ld,
                                     int ncols, float* __restrict__ rowmax,
                                     float* __restrict__ rowsum) {
  __shared__ float red[256];
  const int tid = threadIdx.x;
  const float* row = l + (size_t)blockIdx.x * ld;
  float mx = -3.0e38f;
  for (int c = tid; c < ncols; c += 256) mx = fmaxf(mx, row[c]);
  red[tid] = mx;
  __syncthreads();
  for (int s = 128; s >= 1; s >>= 1) {
    if (tid < s) red[tid] = fmaxf(red[tid], red[tid + s]);
    __syncthreads();
  }
  mx = red[0];
  __syncthreads();
  float sm = 0.0f;
  for (int c = tid; c < ncols; c += 256) sm += __expf(row[c] - mx);
  red[tid] = sm;
  __syncthreads();
  for (int s = 128; s >= 1; s >>= 1) {
    if (tid < s) red[tid] += red[tid + s];
    __syncthreads();
  }
  if (tid == 0) {
    rowmax[blockIdx.x] = mx;
    rowsum[blockIdx.x] = red[0];
  }
}

// ---------------------------------------------------------------------------
// p = softmax(l) @ E : M=512, N=128, K=ncols. grid=(32,8); 8 waves split K,
// weights exp(l-max)/sum formed on the fly as f16 A-fragments; E rows f16 B.
// ---------------------------------------------------------------------------
__global__ __launch_bounds__(256) void softmax_wsum_kernel(
    const float* __restrict__ l, long ld, int ncols,
    const float* __restrict__ rowmax, const float* __restrict__ rowsum,
    const _Float16* __restrict__ Erows, float* __restrict__ p) {
  __shared__ float red[8 * 256];
  const int tid = threadIdx.x;
  const int wave = tid >> 5;
  const int lane = tid & 31;
  const int m = lane & 15, hq = lane >> 4;
  const int m_base = blockIdx.x * 16;
  const int n_base = blockIdx.y * 16;
  const float mymax = rowmax[m_base + m];
  const float myinv = 1.0f / rowsum[m_base + m];
  const float* lrow = l + (size_t)(m_base + m) * ld;

  v8f acc = {};
  for (int kc = wave * 32; kc < ncols; kc += 256) {
    __builtin_prefetch(&lrow[kc + 256], 0, 1);  // global_prefetch_b8
    v16h af;
#pragma unroll
    for (int i = 0; i < 16; ++i) {
      const int k = (i < 8) ? (hq * 8 + i) : (16 + hq * 8 + (i - 8));
      const int vv = kc + k;
      float w = 0.0f;
      if (vv < ncols) w = __expf(lrow[vv] - mymax) * myinv;
      af[i] = (_Float16)w;
    }
    v16h bf;
    const int n = lane & 15;
#pragma unroll
    for (int i = 0; i < 16; ++i) {
      const int vv = kc + ((i < 8) ? (hq * 8 + i) : (16 + hq * 8 + (i - 8)));
      bf[i] = (vv < ncols) ? Erows[(size_t)vv * 128 + n_base + n] : (_Float16)0.0f;
    }
    acc = wmma_f16(af, bf, acc);
  }
#pragma unroll
  for (int r = 0; r < 8; ++r) red[wave * 256 + lane * 8 + r] = acc[r];
  __syncthreads();
  // cross-wave reduce: thread -> (lane2, r2)
  const int lane2 = tid >> 3, r2 = tid & 7;
  float s = 0.0f;
#pragma unroll
  for (int w = 0; w < 8; ++w) s += red[w * 256 + lane2 * 8 + r2];
  const int mm = r2 + 8 * (lane2 >> 4);
  const int nn = lane2 & 15;
  p[(size_t)(m_base + mm) * 128 + n_base + nn] = s;
}

// ---------------------------------------------------------------------------
// coef[b,c,dd] = sum_{j<128} p[b,j]*Wc[c,j,dd] + sum_{j<8} v[b,c,j]*Wc[c,128+j,dd]
// ---------------------------------------------------------------------------
__global__ void coef_kernel(const float* __restrict__ p,
                            const float* __restrict__ v,
                            const float* __restrict__ Wc,
                            float* __restrict__ coef) {
  const size_t idx = (size_t)blockIdx.x * blockDim.x + threadIdx.x;  // B*C*8
  const int b = (int)(idx >> 10);
  const int c = (int)((idx >> 3) & 127);
  const int dd = (int)(idx & 7);
  const float* pr = p + (size_t)b * 128;
  const float* wc = Wc + (size_t)c * 136 * CAPD + dd;
  float s = 0.0f;
#pragma unroll 8
  for (int j = 0; j < 128; ++j) s += pr[j] * wc[(size_t)j * CAPD];
  const float* vv = v + ((size_t)b * 128 + c) * CAPD;
#pragma unroll
  for (int j = 0; j < 8; ++j) s += vv[j] * wc[(size_t)(128 + j) * CAPD];
  coef[idx] = s;
}

// ---------------------------------------------------------------------------
// b[b,h,c] += sum_d u[b,h,c,d] * coef[b,c,d]
// ---------------------------------------------------------------------------
__global__ void bupdate_kernel(const float* __restrict__ u,
                               const float* __restrict__ coef,
                               float* __restrict__ bL) {
  const size_t idx = (size_t)blockIdx.x * blockDim.x + threadIdx.x;  // B*H*C
  const int b = (int)(idx >> 14);
  const int c = (int)(idx & 127);
  const float* uu = u + idx * CAPD;
  const float* cf = coef + ((size_t)b * 128 + c) * CAPD;
  float s = 0.0f;
#pragma unroll
  for (int d = 0; d < CAPD; ++d) s += uu[d] * cf[d];
  bL[idx] += s;
}

// ---------------------------------------------------------------------------
// Host launcher
// ---------------------------------------------------------------------------
extern "C" void kernel_launch(void* const* d_in, const int* in_sizes, int n_in,
                              void* d_out, int out_size, void* d_ws, size_t ws_size,
                              hipStream_t stream) {
  (void)in_sizes; (void)n_in; (void)out_size; (void)ws_size;
  const int* seqs[NBEH] = {(const int*)d_in[0], (const int*)d_in[1],
                           (const int*)d_in[2], (const int*)d_in[3]};
  const int Ts[NBEH] = {50, 50, 50, 200};
  const float* item_emb = (const float*)d_in[4];
  const float* gru_Wih  = (const float*)d_in[5];   // (4,384,128)
  const float* gru_Whh  = (const float*)d_in[6];
  const float* gru_bih  = (const float*)d_in[7];   // (4,384)
  const float* gru_bhh  = (const float*)d_in[8];
  const float* W        = (const float*)d_in[9];   // (128,4,128,8)
  const float* w_scale  = (const float*)d_in[10];
  const float* bias_vec = (const float*)d_in[11];
  const float* alpha    = (const float*)d_in[12];
  const float* W_coef   = (const float*)d_in[13];  // (128,136,8)
  float* out = (float*)d_out;                      // (512, 50001)

  // -- workspace carve-up (256B aligned) --
  char* base = (char*)d_ws;
  size_t off = 0;
  auto carve = [&](size_t bytes) -> char* {
    char* p = base + off;
    off = (off + bytes + 255) & ~(size_t)255;
    return p;
  };
  _Float16* itemE16 = (_Float16*)carve((size_t)VPAD * 128 * 2);
  _Float16* wih16   = (_Float16*)carve((size_t)NBEH * GATES * 128 * 2);
  _Float16* whh16   = (_Float16*)carve((size_t)NBEH * GATES * 128 * 2);
  float*    e_buf   = (float*)carve((size_t)BATCH * 128 * NBEH * 4);
  float*    u_buf   = (float*)carve((size_t)BATCH * 128 * 128 * CAPD * 4);
  float*    bL      = (float*)carve((size_t)BATCH * 128 * 128 * 4);
  float*    cC      = (float*)carve((size_t)BATCH * 128 * 128 * 4);
  float*    v_buf   = (float*)carve((size_t)BATCH * 128 * CAPD * 4);
  float*    vnorm   = (float*)carve((size_t)BATCH * 128 * 4);
  _Float16* vnorm16 = (_Float16*)carve((size_t)BATCH * 128 * 2);
  float*    l_buf   = (float*)carve((size_t)BATCH * VPAD * 4);
  float*    rowmax  = (float*)carve((size_t)BATCH * 4);
  float*    rowsum  = (float*)carve((size_t)BATCH * 4);
  float*    p_buf   = (float*)carve((size_t)BATCH * 128 * 4);
  float*    coef    = (float*)carve((size_t)BATCH * 128 * CAPD * 4);

  // -- f16 conversions --
  cvt_itememb_kernel<<<(VPAD * 128) / 256, 256, 0, stream>>>(item_emb, itemE16);
  {
    const int nw = NBEH * GATES * 128;
    cvt_f32_to_f16_kernel<<<(nw + 255) / 256, 256, 0, stream>>>(gru_Wih, wih16, nw);
    cvt_f32_to_f16_kernel<<<(nw + 255) / 256, 256, 0, stream>>>(gru_Whh, whh16, nw);
  }
  zero_f32_kernel<<<(BATCH * 128 * 128) / 256, 256, 0, stream>>>(bL, BATCH * 128 * 128);

  // -- GRUs (fused gather + WMMA gate GEMMs + recurrence) --
  for (int beh = 0; beh < NBEH; ++beh) {
    gru_kernel<<<BATCH / 16, 256, 0, stream>>>(
        seqs[beh], Ts[beh], itemE16,
        wih16 + (size_t)beh * GATES * 128, whh16 + (size_t)beh * GATES * 128,
        gru_bih + (size_t)beh * GATES, gru_bhh + (size_t)beh * GATES,
        e_buf, beh);
  }

  // -- capsule primary votes u --
  u_kernel<<<(BATCH * 128 * 128) / 256, 256, 0, stream>>>(e_buf, W, u_buf);

  const dim3 gemm_grid(BATCH / 16, (VPAD + 127) / 128);  // 32 x 391
  // -- detached routing iterations (R-1 = 2) --
  for (int it = 0; it < 2; ++it) {
    softmaxC_kernel<<<BATCH * 128, 128, 0, stream>>>(bL, cC);
    v_kernel<<<BATCH * 128, 64, 0, stream>>>(cC, u_buf, alpha, w_scale, bias_vec,
                                             v_buf, vnorm, vnorm16);
    gemm_aBt_kernel<<<gemm_grid, 256, 0, stream>>>(vnorm16, itemE16, l_buf,
                                                   (long)VPAD, VPAD);
    softmax_stats_kernel<<<BATCH, 256, 0, stream>>>(l_buf, (long)VPAD, VOCAB,
                                                    rowmax, rowsum);
    softmax_wsum_kernel<<<dim3(BATCH / 16, 8), 256, 0, stream>>>(
        l_buf, (long)VPAD, VOCAB, rowmax, rowsum, itemE16, p_buf);
    coef_kernel<<<(BATCH * 128 * CAPD) / 256, 256, 0, stream>>>(p_buf, v_buf,
                                                                W_coef, coef);
    bupdate_kernel<<<(BATCH * 128 * 128) / 256, 256, 0, stream>>>(u_buf, coef, bL);
  }

  // -- final pass: logits straight to d_out --
  softmaxC_kernel<<<BATCH * 128, 128, 0, stream>>>(bL, cC);
  v_kernel<<<BATCH * 128, 64, 0, stream>>>(cC, u_buf, alpha, w_scale, bias_vec,
                                           v_buf, vnorm, vnorm16);
  gemm_aBt_kernel<<<gemm_grid, 256, 0, stream>>>(vnorm16, itemE16, out,
                                                 (long)VOCAB, VOCAB);
}
